// FCN8sAtOnceMultiGnn2_69810398429293
// MI455X (gfx1250) — compile-verified, matrix operand order checked by reading, not run on hardware
//
#include <hip/hip_runtime.h>
#include <math.h>

typedef __attribute__((ext_vector_type(16))) _Float16 v16h;
typedef __attribute__((ext_vector_type(8)))  _Float16 v8h;
typedef __attribute__((ext_vector_type(8)))  float    v8f;

#define HW   1024
#define CCH  512
#define NHW  4096   // N*HW
#define KNN  16

// ---------------------------------------------------------------------------
// 2x2 maxpool, NCHW[4,512,64,64] -> node-major [4096 nodes][512 ch]
// ---------------------------------------------------------------------------
__global__ __launch_bounds__(256)
void maxpool_kernel(const float* __restrict__ in, float* __restrict__ out)
{
    const int i   = blockIdx.x * 256 + threadIdx.x;   // 0 .. 2M-1
    const int n   = i >> 19;                          // /(512*1024)
    const int c   = (i >> 10) & 511;
    const int pos = i & 1023;
    const int hh = pos >> 5, ww = pos & 31;
    const float* src = in + (((size_t)(n * 512 + c) * 64) + 2 * hh) * 64 + 2 * ww;
    const float m = fmaxf(fmaxf(src[0], src[1]), fmaxf(src[64], src[65]));
    out[((size_t)(n * 1024 + pos)) * 512 + c] = m;
}

// ---------------------------------------------------------------------------
// W [1024,512] fp32 -> Wt [512][1024] f16 (B-fragment friendly layout)
// ---------------------------------------------------------------------------
__global__ __launch_bounds__(256)
void transw_kernel(const float* __restrict__ w, _Float16* __restrict__ wt)
{
    const int i  = blockIdx.x * 256 + threadIdx.x;    // 0 .. 524287
    const int cc = i >> 10;
    const int kk = i & 1023;
    wt[i] = (_Float16)w[(size_t)kk * 512 + cc];
}

// ---------------------------------------------------------------------------
// Row-normalize (F.normalize, eps 1e-12) and convert to f16. One wave per row.
// ---------------------------------------------------------------------------
__global__ __launch_bounds__(256)
void normalize_kernel(const float* __restrict__ h, _Float16* __restrict__ xn)
{
    const int lane = threadIdx.x & 31;
    const int row  = blockIdx.x * 8 + (threadIdx.x >> 5);
    const float* hr = h + (size_t)row * 512;
    float ss = 0.f;
#pragma unroll
    for (int i = 0; i < 16; ++i) { const float v = hr[lane + 32 * i]; ss += v * v; }
#pragma unroll
    for (int d = 16; d >= 1; d >>= 1) ss += __shfl_xor(ss, d);
    const float inv = 1.0f / fmaxf(sqrtf(ss), 1e-12f);
    _Float16* xr = xn + (size_t)row * 512;
#pragma unroll
    for (int i = 0; i < 16; ++i) xr[lane + 32 * i] = (_Float16)(hr[lane + 32 * i] * inv);
}

// ---------------------------------------------------------------------------
// Per-sample Gram matrix S = Xn * Xn^T via WMMA f16 -> f32.
// Rows normalized => knn(smallest d2) == argmax-16 of S.
// grid (16,16,4): tileN, tileM, sample. 64x64 tile, 8 waves, 2 frags/wave.
// ---------------------------------------------------------------------------
__global__ __launch_bounds__(256)
void gram_kernel(const _Float16* __restrict__ xn, float* __restrict__ S)
{
    const int tid  = threadIdx.x;
    const int lane = tid & 31, wave = tid >> 5;
    const int Mblk = wave >> 1, nhalf = wave & 1;
    const _Float16* x = xn + (size_t)blockIdx.z * HW * CCH;
    float* Ss = S + (size_t)blockIdx.z * HW * HW;

    const int mrow  = blockIdx.y * 64 + Mblk * 16 + (lane & 15);
    const int ncol0 = blockIdx.x * 64 + nhalf * 32 + (lane & 15);
    const int akh   = (lane >> 4) * 16;

    v8f acc0 = {}, acc1 = {};
    for (int kk = 0; kk < 16; ++kk) {
        const int kc = kk * 32 + akh;
        const v16h a  = *(const v16h*)(x + (size_t)mrow * 512 + kc);
        const v16h b0 = *(const v16h*)(x + (size_t)ncol0 * 512 + kc);
        const v16h b1 = *(const v16h*)(x + (size_t)(ncol0 + 16) * 512 + kc);
        acc0 = __builtin_amdgcn_wmma_f32_16x16x32_f16(false, a, false, b0, (short)0, acc0, false, false);
        acc1 = __builtin_amdgcn_wmma_f32_16x16x32_f16(false, a, false, b1, (short)0, acc1, false, false);
    }
    const int rbase = blockIdx.y * 64 + Mblk * 16 + 8 * (lane >> 4);
#pragma unroll
    for (int g = 0; g < 8; ++g) {
        const int r = rbase + g;
        Ss[(size_t)r * 1024 + ncol0]      = acc0[g];
        Ss[(size_t)r * 1024 + ncol0 + 16] = acc1[g];
    }
}

// ---------------------------------------------------------------------------
// Top-16 largest similarities per row (== 16 nearest). Register insertion.
// Strict '>' keeps the earlier index on ties (top_k tie order).
// ---------------------------------------------------------------------------
__global__ __launch_bounds__(256)
void topk_kernel(const float* __restrict__ S, int* __restrict__ idx)
{
    const int r    = blockIdx.x * 256 + threadIdx.x;  // 0..4095
    const int samp = r >> 10;
    const int i    = r & 1023;
    const float* row = S + (size_t)samp * HW * HW + (size_t)i * 1024;
    float val[16]; int id[16];
#pragma unroll
    for (int t = 0; t < 16; ++t) { val[t] = -1e30f; id[t] = 0; }
    for (int j = 0; j < 1024; ++j) {
        float v = row[j];
        int cid = j;
#pragma unroll
        for (int t = 0; t < 16; ++t) {
            if (v > val[t]) {
                const float tv = val[t]; val[t] = v; v = tv;
                const int   ti = id[t];  id[t] = cid; cid = ti;
            }
        }
    }
#pragma unroll
    for (int t = 0; t < 16; ++t) idx[(size_t)r * 16 + t] = id[t];
}

// ---------------------------------------------------------------------------
// Fused GNN layer: gather neighbors, build [nb, nb - nb_other] f16 in LDS,
// GEMM vs Wt with WMMA, epilogue = bias + leaky-relu + mean over the 16
// neighbors (one WMMA 16-row block == one node's K group).
// grid (8, 1024): tileN (8*64=512 out ch), tileM (1024*64=65536 rows).
// ---------------------------------------------------------------------------
__global__ __launch_bounds__(256)
void gnn_gemm_kernel(const float* __restrict__ h_pri,
                     const float* __restrict__ h_sec,
                     const int* __restrict__ idx_pri,
                     const int* __restrict__ idx_sec,
                     const _Float16* __restrict__ wt,   // [512][1024] f16
                     const float* __restrict__ bias,    // [512]
                     float* __restrict__ m_out)         // [4096][512]
{
    __shared__ int ri[64];
    __shared__ int ii[64];
    __shared__ _Float16 atile[64 * 40];   // stride padded 32->40 halfwords

    const int tid   = threadIdx.x;
    const int tileN = blockIdx.x;
    const int tileM = blockIdx.y;
    const int mBase = tileM * 64;

    if (tid < 64)       ri[tid]      = idx_pri[mBase + tid];
    else if (tid < 128) ii[tid - 64] = idx_sec[mBase + tid - 64];
    __syncthreads();

    const int lane  = tid & 31;
    const int wave  = tid >> 5;
    const int Mblk  = wave >> 1;
    const int nhalf = wave & 1;

    const int grow = tid >> 2;     // gather row 0..63
    const int c0   = (tid & 3) * 8;

    const int arow  = Mblk * 16 + (lane & 15);
    const int akh   = (lane >> 4) * 16;
    const int ncol0 = tileN * 64 + nhalf * 32 + (lane & 15);
    const _Float16* wb0 = wt + (size_t)ncol0 * 1024;
    const _Float16* wb1 = wt + (size_t)(ncol0 + 16) * 1024;

    v8f acc0 = {}, acc1 = {};

    for (int kk = 0; kk < 32; ++kk) {
        const int kc = kk * 32;
        // ---- gather + convert: 64 rows x 32 f16 into LDS
        {
            const int cg = kc + c0;
            const int rp = ri[grow];
            float v[8];
            if (cg < 512) {
                const float4 a = *(const float4*)(h_pri + (size_t)rp * 512 + cg);
                const float4 b = *(const float4*)(h_pri + (size_t)rp * 512 + cg + 4);
                v[0]=a.x; v[1]=a.y; v[2]=a.z; v[3]=a.w;
                v[4]=b.x; v[5]=b.y; v[6]=b.z; v[7]=b.w;
            } else {
                const int cs = cg - 512;
                const int rs = ii[grow];
                const float4 a = *(const float4*)(h_pri + (size_t)rp * 512 + cs);
                const float4 b = *(const float4*)(h_pri + (size_t)rp * 512 + cs + 4);
                const float4 c = *(const float4*)(h_sec + (size_t)rs * 512 + cs);
                const float4 d = *(const float4*)(h_sec + (size_t)rs * 512 + cs + 4);
                v[0]=a.x-c.x; v[1]=a.y-c.y; v[2]=a.z-c.z; v[3]=a.w-c.w;
                v[4]=b.x-d.x; v[5]=b.y-d.y; v[6]=b.z-d.z; v[7]=b.w-d.w;
            }
            v8h hv;
#pragma unroll
            for (int e = 0; e < 8; ++e) hv[e] = (_Float16)v[e];
            *(v8h*)(&atile[grow * 40 + c0]) = hv;
        }
        __syncthreads();

        union { v16h v; v8h h[2]; } au;
        au.h[0] = *(const v8h*)(&atile[arow * 40 + akh]);
        au.h[1] = *(const v8h*)(&atile[arow * 40 + akh + 8]);

        const v16h b0 = *(const v16h*)(wb0 + kc + akh);
        const v16h b1 = *(const v16h*)(wb1 + kc + akh);

        acc0 = __builtin_amdgcn_wmma_f32_16x16x32_f16(false, au.v, false, b0, (short)0, acc0, false, false);
        acc1 = __builtin_amdgcn_wmma_f32_16x16x32_f16(false, au.v, false, b1, (short)0, acc1, false, false);
        __syncthreads();
    }

    // ---- epilogue: bias + leaky-relu, mean over the 16 neighbor rows
    const int   p   = tileM * 4 + Mblk;
    const float b0v = bias[ncol0];
    const float b1v = bias[ncol0 + 16];
    float s0 = 0.f, s1 = 0.f;
#pragma unroll
    for (int g = 0; g < 8; ++g) {
        const float x0 = acc0[g] + b0v;
        const float x1 = acc1[g] + b1v;
        s0 += (x0 >= 0.f) ? x0 : 0.01f * x0;
        s1 += (x1 >= 0.f) ? x1 : 0.01f * x1;
    }
    s0 += __shfl_xor(s0, 16);
    s1 += __shfl_xor(s1, 16);
    if (lane < 16) {
        m_out[(size_t)p * 512 + ncol0]      = s0 * (1.0f / 16.0f);
        m_out[(size_t)p * 512 + ncol0 + 16] = s1 * (1.0f / 16.0f);
    }
}

// ---------------------------------------------------------------------------
// Per-sample SE gate: mean over HW, 1024->32 lrelu, 32->512 sigmoid.
// ---------------------------------------------------------------------------
__global__ __launch_bounds__(256)
void se_kernel(const float* __restrict__ m_r, const float* __restrict__ m_i,
               const float* __restrict__ w1, const float* __restrict__ b1,
               const float* __restrict__ w2, const float* __restrict__ b2,
               float* __restrict__ gate)
{
    __shared__ float cc[1024];
    __shared__ float zz[32];
    const int s   = blockIdx.x;
    const int tid = threadIdx.x;
    for (int half = 0; half < 2; ++half) {
        const int ch = tid + half * 256;
        float ar = 0.f, ai = 0.f;
        const float* pr = m_r + ((size_t)s * 1024) * 512 + ch;
        const float* pi = m_i + ((size_t)s * 1024) * 512 + ch;
        for (int r = 0; r < 1024; ++r) { ar += pr[(size_t)r * 512]; ai += pi[(size_t)r * 512]; }
        cc[ch]       = ar * (1.f / 1024.f);
        cc[512 + ch] = ai * (1.f / 1024.f);
    }
    __syncthreads();
    if (tid < 32) {
        float z = b1[tid];
        for (int c = 0; c < 1024; ++c) z += cc[c] * w1[(size_t)c * 32 + tid];
        zz[tid] = (z >= 0.f) ? z : 0.01f * z;
    }
    __syncthreads();
    for (int half = 0; half < 2; ++half) {
        const int ch = tid + half * 256;
        float g = b2[ch];
#pragma unroll
        for (int j = 0; j < 32; ++j) g += zz[j] * w2[(size_t)j * 512 + ch];
        gate[(size_t)s * 512 + ch] = 1.0f / (1.0f + expf(-g));
    }
}

// ---------------------------------------------------------------------------
// h_rgb *= gate ; h_ir *= (1-gate) ; last iter: out = relu(g1*hr+g2*hi) NCHW
// ---------------------------------------------------------------------------
__global__ __launch_bounds__(256)
void update_kernel(float* __restrict__ h_rgb, float* __restrict__ h_ir,
                   const float* __restrict__ gate,
                   const float* __restrict__ g1, const float* __restrict__ g2,
                   float* __restrict__ out, int write_out)
{
    const int e   = blockIdx.x * 256 + threadIdx.x;   // 0..2M-1
    const int p   = e >> 9;
    const int c   = e & 511;
    const int s   = p >> 10;
    const int pos = p & 1023;
    const float ga = gate[s * 512 + c];
    const float hr = h_rgb[e] * ga;
    const float hi = h_ir[e] * (1.0f - ga);
    h_rgb[e] = hr;
    h_ir[e]  = hi;
    if (write_out) {
        const float v = g1[0] * hr + g2[0] * hi;
        out[((size_t)(s * 512 + c) << 10) + pos] = fmaxf(v, 0.0f);
    }
}

// ---------------------------------------------------------------------------
extern "C" void kernel_launch(void* const* d_in, const int* in_sizes, int n_in,
                              void* d_out, int out_size, void* d_ws, size_t ws_size,
                              hipStream_t stream)
{
    (void)in_sizes; (void)n_in; (void)out_size; (void)ws_size;
    const float* rgb    = (const float*)d_in[0];
    const float* ir     = (const float*)d_in[1];
    const float* W_rgb  = (const float*)d_in[2];
    const float* b_rgb  = (const float*)d_in[3];
    const float* W_ir   = (const float*)d_in[4];
    const float* b_ir   = (const float*)d_in[5];
    const float* W_se1  = (const float*)d_in[6];
    const float* b_se1  = (const float*)d_in[7];
    const float* W_se2  = (const float*)d_in[8];
    const float* b_se2  = (const float*)d_in[9];
    const float* gamma1 = (const float*)d_in[10];
    const float* gamma2 = (const float*)d_in[11];
    float* out = (float*)d_out;

    char* ws = (char*)d_ws;
    float*    HR   = (float*)(ws);                               // 8 MiB
    float*    HI   = (float*)(ws + ((size_t)8  << 20));          // 8 MiB
    _Float16* WTR  = (_Float16*)(ws + ((size_t)16 << 20));       // 1 MiB
    _Float16* WTI  = (_Float16*)(ws + ((size_t)17 << 20));       // 1 MiB
    int*      IDXR = (int*)(ws + ((size_t)18 << 20));            // 256 KiB
    int*      IDXI = (int*)(ws + ((size_t)18 << 20) + (256u << 10));
    float*    GATE = (float*)(ws + ((size_t)18 << 20) + (512u << 10));
    char*     SCR  = ws + ((size_t)19 << 20);                    // 24 MiB union
    _Float16* XNR  = (_Float16*)(SCR);                           // 4 MiB
    _Float16* XNI  = (_Float16*)(SCR + ((size_t)4 << 20));       // 4 MiB
    float*    S    = (float*)(SCR + ((size_t)8 << 20));          // 16 MiB
    float*    MR   = (float*)(SCR);                              // 8 MiB (after knn)
    float*    MI   = (float*)(SCR + ((size_t)8 << 20));          // 8 MiB (after knn)

    maxpool_kernel<<<8192, 256, 0, stream>>>(rgb, HR);
    maxpool_kernel<<<8192, 256, 0, stream>>>(ir,  HI);
    transw_kernel<<<2048, 256, 0, stream>>>(W_rgb, WTR);
    transw_kernel<<<2048, 256, 0, stream>>>(W_ir,  WTI);
    normalize_kernel<<<512, 256, 0, stream>>>(HR, XNR);
    normalize_kernel<<<512, 256, 0, stream>>>(HI, XNI);

    gram_kernel<<<dim3(16, 16, 4), 256, 0, stream>>>(XNR, S);
    topk_kernel<<<16, 256, 0, stream>>>(S, IDXR);
    gram_kernel<<<dim3(16, 16, 4), 256, 0, stream>>>(XNI, S);
    topk_kernel<<<16, 256, 0, stream>>>(S, IDXI);

    for (int it = 0; it < 3; ++it) {   // gnn_iterations = 3 (setup_inputs)
        gnn_gemm_kernel<<<dim3(8, 1024), 256, 0, stream>>>(HR, HI, IDXR, IDXI, WTR, b_rgb, MR);
        gnn_gemm_kernel<<<dim3(8, 1024), 256, 0, stream>>>(HI, HR, IDXI, IDXR, WTI, b_ir,  MI);
        se_kernel<<<4, 256, 0, stream>>>(MR, MI, W_se1, b_se1, W_se2, b_se2, GATE);
        update_kernel<<<8192, 256, 0, stream>>>(HR, HI, GATE, gamma1, gamma2, out, it == 2 ? 1 : 0);
    }
}
